// DepthDeformConvPack_40905268527561
// MI455X (gfx1250) — matrix operand order repaired and checked by reading, not compiled, not run on hardware
//
#include <hip/hip_runtime.h>
#include <hip/hip_bf16.h>
#include <math.h>

// ---------------- problem constants ----------------
#define NB    4
#define HH    64
#define WW    64
#define PP    (HH*WW)          // 4096 pixels per image
#define CIN   256
#define DC    64
#define COUT  256
#define KT    9                // 3x3 taps
#define KDIM  (CIN*KT)         // 2304 GEMM K (main conv)
#define OFFC  18               // 2*KT offset channels
#define OCIN  (CIN+DC)         // 320: offset-conv input channels

#define KITERS 72              // main conv: 9 taps * (256/32)
#define KIT1   90              // offset conv: 9 taps * (320/32)
#define ASTR   40              // padded LDS row stride (bf16)

typedef __attribute__((ext_vector_type(16))) __bf16 v16bf;
typedef __attribute__((ext_vector_type(8)))  __bf16 v8bf;
typedef __attribute__((ext_vector_type(8)))  float  v8f;
typedef __attribute__((ext_vector_type(4)))  float  v4f;

static __device__ __forceinline__ unsigned short f2bfbits(float f) {
    union { float f; unsigned u; } in; in.f = f;
    unsigned u = in.u;
    return (unsigned short)((u + 0x7FFFu + ((u >> 16) & 1u)) >> 16);
}
static __device__ __forceinline__ unsigned pack2bf(float a, float b) {
    return (unsigned)f2bfbits(a) | ((unsigned)f2bfbits(b) << 16);
}

#define SHUF16(lo, hi) __builtin_shufflevector(lo, hi, 0,1,2,3,4,5,6,7,8,9,10,11,12,13,14,15)

// ---------------- NCHW -> NHWC ----------------
__global__ void k_nchw_to_nhwc(const float* __restrict__ in, float* __restrict__ out,
                               int C, int total) {
    int i = blockIdx.x * 256 + threadIdx.x;
    if (i >= total) return;
    int c = i % C;
    int r = i / C;
    int p = r % PP;
    int n = r / PP;
    out[i] = in[((size_t)n * C + c) * PP + p];
}

// ---------------- main weight -> fragment-ready bf16 ----------------
// B frag (16x16x32): lane l = column N=l%16; lanes 0-15 K=0..15, 16-31 K=16..31.
// K' = k*256 + c. Stream: frag[nb(16)][kiter(72)][lane(32)][16 halves]
__global__ void k_weight_frag(const float* __restrict__ w, unsigned short* __restrict__ wf) {
    int i = blockIdx.x * 256 + threadIdx.x;          // 16*72*32
    if (i >= 16 * KITERS * 32) return;
    int lane  = i & 31;
    int kiter = (i >> 5) % KITERS;
    int nb    = i / (KITERS * 32);
    int k  = kiter >> 3;
    int c0 = (kiter & 7) * 32;
    int ncol  = nb * 16 + (lane & 15);
    int kbase = (lane < 16) ? 0 : 16;
    unsigned short* dst = wf + (size_t)i * 16;
#pragma unroll
    for (int h = 0; h < 16; ++h) {
        int c = c0 + kbase + h;
        dst[h] = f2bfbits(w[(size_t)ncol * KDIM + c * KT + k]);
    }
}

// ---------------- offset-conv weight -> fragment-ready bf16 ----------------
// K' = k*320 + c ; N padded 18 -> 32 with zeros. Stream: [nb(2)][kit(90)][lane][16]
__global__ void k_offw_frag(const float* __restrict__ ow, unsigned short* __restrict__ wf) {
    int i = blockIdx.x * 256 + threadIdx.x;          // 2*90*32 = 5760
    if (i >= 2 * KIT1 * 32) return;
    int lane  = i & 31;
    int kiter = (i >> 5) % KIT1;
    int nb    = i / (KIT1 * 32);
    int k     = kiter / 10;
    int c0    = (kiter - k * 10) * 32;
    int ncol  = nb * 16 + (lane & 15);
    int kbase = (lane < 16) ? 0 : 16;
    unsigned short* dst = wf + (size_t)i * 16;
#pragma unroll
    for (int h = 0; h < 16; ++h) {
        int c = c0 + kbase + h;
        float v = (ncol < OFFC) ? ow[(size_t)(ncol * OCIN + c) * KT + k] : 0.f;
        dst[h] = f2bfbits(v);
    }
}

// ---------------- kernel 1: offset conv via WMMA ----------------
// Block = 128 pixels, 8 waves; wave w owns M rows [w*16, w*16+16), N = 32 (2 blocks).
__global__ void __launch_bounds__(256)
k_offset_wmma(const float* __restrict__ xh, const float* __restrict__ dh,
              const unsigned short* __restrict__ wf, const float* __restrict__ ob,
              float* __restrict__ offs) {
    __shared__ int    Sbase[128 * KT];
    __shared__ __bf16 Ash[2][128 * ASTR];

    const int t    = threadIdx.x;
    const int lane = t & 31;
    const int wave = t >> 5;
    const int tile = blockIdx.x;                  // 128 tiles of 128 pixels
    const int img  = (tile * 128) >> 12;
    const int ptl  = (tile * 128) & (PP - 1);

    for (int idx = t; idx < 128 * KT; idx += 256) {
        int m = idx / KT, k = idx - m * KT;
        int p = ptl + m;
        int y = p >> 6, x = p & 63;
        int yy = y + (k / 3) - 1, xx = x + (k % 3) - 1;
        bool valid = (yy >= 0) & (yy < HH) & (xx >= 0) & (xx < WW);
        Sbase[idx] = valid ? (img * PP + yy * WW + xx) : -1;
    }
    __syncthreads();

    auto buildA = [&](int it, int buf) {
        int k     = it / 10;
        int chunk = it - k * 10;
        int c0    = chunk * 32;
        for (int q = t; q < 1024; q += 256) {     // 128 rows x 8 quads
            int m  = q >> 3;
            int cq = (q & 7) * 4;
            int pos = Sbase[m * KT + k];
            v4f v = {};
            if (pos >= 0) {
                if (chunk < 8) v = *(const v4f*)(xh + (size_t)pos * CIN + c0 + cq);
                else           v = *(const v4f*)(dh + (size_t)pos * DC + (c0 - CIN) + cq);
            }
            *(uint2*)&Ash[buf][m * ASTR + cq] =
                make_uint2(pack2bf(v[0], v[1]), pack2bf(v[2], v[3]));
        }
    };

    v8f acc0 = {}, acc1 = {};
    const int lm = lane & 15, hi = lane >> 4;
    buildA(0, 0);
    for (int it = 0; it < KIT1; ++it) {
        __syncthreads();
        int buf = it & 1;
        const __bf16* arow = &Ash[buf][(wave * 16 + lm) * ASTR];
        v8bf alo = *(const v8bf*)(arow + (hi ? 16 : 0));
        v8bf ahi = *(const v8bf*)(arow + (hi ? 24 : 8));
        const v8bf* b0p = (const v8bf*)(wf + ((size_t)(0 * KIT1 + it) * 32 + lane) * 16);
        const v8bf* b1p = (const v8bf*)(wf + ((size_t)(1 * KIT1 + it) * 32 + lane) * 16);
        v16bf af  = SHUF16(alo, ahi);
        v16bf bf0 = SHUF16(b0p[0], b0p[1]);
        v16bf bf1 = SHUF16(b1p[0], b1p[1]);
        if (it + 1 < KIT1) buildA(it + 1, buf ^ 1);
        acc0 = __builtin_amdgcn_wmma_f32_16x16x32_bf16(false, af, false, bf0, (short)0, acc0, false, false);
        acc1 = __builtin_amdgcn_wmma_f32_16x16x32_bf16(false, af, false, bf1, (short)0, acc1, false, false);
    }

    // epilogue: N = lm (acc0) and 16+lm (acc1, only lm<2 valid); M = wave*16 + r + 8*hi
    float b0 = ob[lm];
    float b1 = (lm < 2) ? ob[16 + lm] : 0.f;
#pragma unroll
    for (int r = 0; r < 8; ++r) {
        int p = ptl + wave * 16 + r + hi * 8;
        offs[((size_t)img * OFFC + lm) * PP + p] = acc0[r] + b0;
        if (lm < 2)
            offs[((size_t)img * OFFC + 16 + lm) * PP + p] = acc1[r] + b1;
    }
}

// ---------------- kernel 2: mask = sigmoid(deform_conv(depth)) ----------------
__global__ void k_mask(const float* __restrict__ dh, const float* __restrict__ offs,
                       const float* __restrict__ mw, const float* __restrict__ mb,
                       float* __restrict__ msk) {
    int pix = blockIdx.x * 256 + threadIdx.x;
    if (pix >= NB * PP) return;
    int n = pix >> 12, p = pix & (PP - 1);
    int y = p >> 6, x = p & 63;
    float acc[KT];
#pragma unroll
    for (int o = 0; o < KT; ++o) acc[o] = mb[o];
    const float* offp = offs + (size_t)n * OFFC * PP;
    for (int k = 0; k < KT; ++k) {
        float dy = offp[(2 * k + 0) * PP + p];
        float dx = offp[(2 * k + 1) * PP + p];
        float py = (float)(y + (k / 3) - 1) + dy;
        float px = (float)(x + (k % 3) - 1) + dx;
        float fy = floorf(py), fx = floorf(px);
        float ly = py - fy, lx = px - fx;
        int y0 = (int)fy, x0 = (int)fx;
        int base[4]; float wt[4];
#pragma unroll
        for (int nbr = 0; nbr < 4; ++nbr) {
            int yy = y0 + (nbr >> 1), xx = x0 + (nbr & 1);
            bool valid = (yy >= 0) & (yy < HH) & (xx >= 0) & (xx < WW);
            float w = ((nbr >> 1) ? ly : 1.f - ly) * ((nbr & 1) ? lx : 1.f - lx);
            wt[nbr] = valid ? w : 0.f;
            int yc = min(max(yy, 0), HH - 1), xc = min(max(xx, 0), WW - 1);
            base[nbr] = (n * PP + yc * WW + xc) * DC;
        }
        for (int c = 0; c < DC; ++c) {
            float v = wt[0] * dh[base[0] + c] + wt[1] * dh[base[1] + c]
                    + wt[2] * dh[base[2] + c] + wt[3] * dh[base[3] + c];
#pragma unroll
            for (int o = 0; o < KT; ++o)
                acc[o] += v * mw[(size_t)o * (DC * KT) + c * KT + k];
        }
    }
#pragma unroll
    for (int o = 0; o < KT; ++o)
        msk[((size_t)n * KT + o) * PP + p] = 1.f / (1.f + expf(-acc[o]));
}

// ---------------- kernel 3: main modulated deform conv via WMMA ----------------
// Block = 32 pixels (2 M blocks) x 256 couts; wave w -> couts [w*32,w*32+32);
// 4 WMMAs/wave/iter, double-buffered LDS A tile, 1 barrier/iter.
__global__ void __launch_bounds__(256)
k_deform_wmma(const float* __restrict__ xh, const float* __restrict__ offs,
              const float* __restrict__ msk, const unsigned short* __restrict__ wf,
              const float* __restrict__ bias, float* __restrict__ out) {
    __shared__ int    Sofs[32][KT][4];
    __shared__ float  Swt[32][KT][4];
    __shared__ __bf16 Ash[2][32 * ASTR];

    const int t    = threadIdx.x;
    const int lane = t & 31;
    const int wave = t >> 5;
    const int tile = blockIdx.x;                 // 512 tiles of 32 pixels
    const int img  = (tile * 32) >> 12;
    const int ptl  = (tile * 32) & (PP - 1);

    for (int idx = t; idx < 32 * KT; idx += 256) {
        int m = idx / KT, k = idx - m * KT;
        int p = ptl + m;
        int y = p >> 6, x = p & 63;
        const float* offp = offs + (size_t)img * OFFC * PP;
        float dy = offp[(2 * k + 0) * PP + p];
        float dx = offp[(2 * k + 1) * PP + p];
        float mv = msk[((size_t)img * KT + k) * PP + p];
        float py = (float)(y + (k / 3) - 1) + dy;
        float px = (float)(x + (k % 3) - 1) + dx;
        float fy = floorf(py), fx = floorf(px);
        float ly = py - fy, lx = px - fx;
        int y0 = (int)fy, x0 = (int)fx;
#pragma unroll
        for (int nbr = 0; nbr < 4; ++nbr) {
            int yy = y0 + (nbr >> 1), xx = x0 + (nbr & 1);
            bool valid = (yy >= 0) & (yy < HH) & (xx >= 0) & (xx < WW);
            float w = ((nbr >> 1) ? ly : 1.f - ly) * ((nbr & 1) ? lx : 1.f - lx);
            Swt[m][k][nbr]  = valid ? (w * mv) : 0.f;
            int yc = min(max(yy, 0), HH - 1), xc = min(max(xx, 0), WW - 1);
            Sofs[m][k][nbr] = (img * PP + yc * WW + xc) * CIN;
        }
    }
    __syncthreads();

    auto buildA = [&](int it, int buf) {         // 32 rows x 8 quads = 256 -> 1/thread
        int k  = it >> 3;
        int c0 = (it & 7) * 32;
        int m  = t >> 3;
        int cq = (t & 7) * 4;
        const int*   o4 = &Sofs[m][k][0];
        const float* w4 = &Swt[m][k][0];
        int cc = c0 + cq;
        v4f v = w4[0] * *(const v4f*)(xh + o4[0] + cc)
              + w4[1] * *(const v4f*)(xh + o4[1] + cc)
              + w4[2] * *(const v4f*)(xh + o4[2] + cc)
              + w4[3] * *(const v4f*)(xh + o4[3] + cc);
        *(uint2*)&Ash[buf][m * ASTR + cq] =
            make_uint2(pack2bf(v[0], v[1]), pack2bf(v[2], v[3]));
    };

    v8f acc00 = {}, acc01 = {}, acc10 = {}, acc11 = {};
    const int nb0 = wave * 2, nb1 = wave * 2 + 1;
    const int lm = lane & 15, hi = lane >> 4;

    buildA(0, 0);
    for (int it = 0; it < KITERS; ++it) {
        __syncthreads();
        int buf = it & 1;
        const __bf16* r0 = &Ash[buf][lm * ASTR];
        const __bf16* r1 = &Ash[buf][(16 + lm) * ASTR];
        v8bf a0l = *(const v8bf*)(r0 + (hi ? 16 : 0));
        v8bf a0h = *(const v8bf*)(r0 + (hi ? 24 : 8));
        v8bf a1l = *(const v8bf*)(r1 + (hi ? 16 : 0));
        v8bf a1h = *(const v8bf*)(r1 + (hi ? 24 : 8));
        const v8bf* b0p = (const v8bf*)(wf + ((size_t)(nb0 * KITERS + it) * 32 + lane) * 16);
        const v8bf* b1p = (const v8bf*)(wf + ((size_t)(nb1 * KITERS + it) * 32 + lane) * 16);
        v16bf a0 = SHUF16(a0l, a0h);
        v16bf a1 = SHUF16(a1l, a1h);
        v16bf b0 = SHUF16(b0p[0], b0p[1]);
        v16bf b1 = SHUF16(b1p[0], b1p[1]);
        if (it + 1 < KITERS) {
            __builtin_prefetch(wf + ((size_t)(nb0 * KITERS + it + 1) * 32 + lane) * 16, 0, 1);
            buildA(it + 1, buf ^ 1);
        }
        acc00 = __builtin_amdgcn_wmma_f32_16x16x32_bf16(false, a0, false, b0, (short)0, acc00, false, false);
        acc01 = __builtin_amdgcn_wmma_f32_16x16x32_bf16(false, a0, false, b1, (short)0, acc01, false, false);
        acc10 = __builtin_amdgcn_wmma_f32_16x16x32_bf16(false, a1, false, b0, (short)0, acc10, false, false);
        acc11 = __builtin_amdgcn_wmma_f32_16x16x32_bf16(false, a1, false, b1, (short)0, acc11, false, false);
    }

    {
        int n0 = nb0 * 16 + lm;
        int n1 = nb1 * 16 + lm;
        float b0 = bias[n0], b1 = bias[n1];
        size_t obase = (size_t)img * COUT * PP;
#pragma unroll
        for (int r = 0; r < 8; ++r) {
            int pA = ptl + r + hi * 8;           // M block 0
            int pB = pA + 16;                    // M block 1
            out[obase + (size_t)n0 * PP + pA] = acc00[r] + b0;
            out[obase + (size_t)n1 * PP + pA] = acc01[r] + b1;
            out[obase + (size_t)n0 * PP + pB] = acc10[r] + b0;
            out[obase + (size_t)n1 * PP + pB] = acc11[r] + b1;
        }
    }
}

// ---------------- launcher ----------------
extern "C" void kernel_launch(void* const* d_in, const int* in_sizes, int n_in,
                              void* d_out, int out_size, void* d_ws, size_t ws_size,
                              hipStream_t stream) {
    const float* x      = (const float*)d_in[0];
    const float* depth  = (const float*)d_in[1];
    const float* weight = (const float*)d_in[2];
    const float* bias   = (const float*)d_in[3];
    const float* off_w  = (const float*)d_in[4];
    const float* off_b  = (const float*)d_in[5];
    const float* mask_w = (const float*)d_in[6];
    const float* mask_b = (const float*)d_in[7];
    float* out = (float*)d_out;

    // workspace carve-up (~24.1 MB)
    float* xh   = (float*)d_ws;                                  // 4*4096*256 f32
    float* dh   = xh + (size_t)NB * PP * CIN;                    // 4*4096*64 f32
    float* offs = dh + (size_t)NB * PP * DC;                     // 4*18*4096 f32
    float* mskb = offs + (size_t)NB * OFFC * PP;                 // 4*9*4096 f32
    unsigned short* wfb = (unsigned short*)(mskb + (size_t)NB * KT * PP); // 16*72*32*16 bf16
    unsigned short* wfo = wfb + (size_t)16 * KITERS * 32 * 16;   // 2*90*32*16 bf16

    int totX = NB * CIN * PP;
    int totD = NB * DC * PP;
    k_nchw_to_nhwc<<<(totX + 255) / 256, 256, 0, stream>>>(x, xh, CIN, totX);
    k_nchw_to_nhwc<<<(totD + 255) / 256, 256, 0, stream>>>(depth, dh, DC, totD);
    k_weight_frag<<<(16 * KITERS * 32 + 255) / 256, 256, 0, stream>>>(weight, wfb);
    k_offw_frag<<<(2 * KIT1 * 32 + 255) / 256, 256, 0, stream>>>(off_w, wfo);

    int pix = NB * PP;                    // 16384
    k_offset_wmma<<<pix / 128, 256, 0, stream>>>(xh, dh, wfo, off_b, offs);
    k_mask<<<(pix + 255) / 256, 256, 0, stream>>>(dh, offs, mask_w, mask_b, mskb);
    k_deform_wmma<<<pix / 32, 256, 0, stream>>>(xh, offs, mskb, wfb, bias, out);
}
